// TriangleMultiplication_21766894256696
// MI455X (gfx1250) — matrix operand order repaired and checked by reading, not compiled, not run on hardware
//
#include <hip/hip_runtime.h>

// Triangle multiplication (outgoing), N=768, D=DH=128, fp32 I/O.
// All 3 matmul stages on v_wmma_f32_16x16x32_bf16 (wave32).
// k2 uses GLOBAL_LOAD_ASYNC_TO_LDS double-buffering when available.

#define NN 768
#define DCH 128
constexpr size_t POS = (size_t)NN * NN;   // 589824 positions

typedef __attribute__((ext_vector_type(8)))  __bf16 v8bf;
typedef __attribute__((ext_vector_type(16))) __bf16 v16bf;
typedef __attribute__((ext_vector_type(8)))  float  v8f;
typedef __attribute__((ext_vector_type(4)))  int    v4i;

#if defined(__AMDGCN__) && __has_builtin(__builtin_amdgcn_global_load_async_to_lds_b128) && \
    __has_builtin(__builtin_amdgcn_s_wait_asynccnt)
#define USE_ASYNC_LDS 1
#else
#define USE_ASYNC_LDS 0
#endif

// ---- WMMA fragment helpers (per CDNA5 ISA 7.12.2 layouts) ----
// A 16x32 bf16: lanes 0-15 row=lane, K {0..7,16..23}; lanes 16-31 row=lane-16, K {8..15,24..31}
__device__ __forceinline__ v16bf frag_gap(const __bf16* p) {
  v8bf lo = *(const v8bf*)(p);
  v8bf hi = *(const v8bf*)(p + 16);
  v16bf r;
#pragma unroll
  for (int i = 0; i < 8; ++i) { r[i] = lo[i]; r[i + 8] = hi[i]; }
  return r;
}
// B 32x16 bf16 (k order used here, A same): lanes 0-15 col=lane, K 0..15 contiguous; lanes 16-31 col=lane-16, K 16..31
__device__ __forceinline__ v16bf frag_contig(const __bf16* p) {
  v8bf lo = *(const v8bf*)(p);
  v8bf hi = *(const v8bf*)(p + 8);
  v16bf r;
#pragma unroll
  for (int i = 0; i < 8; ++i) { r[i] = lo[i]; r[i + 8] = hi[i]; }
  return r;
}
__device__ __forceinline__ v16bf afrag(const __bf16* base, int rowBase, int stride, int k0, int lane) {
  int row  = rowBase + (lane & 15);
  int koff = (lane & 16) ? 16 : 0;
  return frag_contig(base + row * stride + k0 + koff);
}
__device__ __forceinline__ v16bf bfrag(const __bf16* base, int rowBase, int stride, int k0, int lane) {
  int row  = rowBase + (lane & 15);
  int koff = (lane & 16) ? 16 : 0;
  return frag_contig(base + row * stride + k0 + koff);
}
__device__ __forceinline__ v8f wmma_bf16(v16bf a, v16bf b, v8f c) {
  // (neg_a, A, neg_b, B, c_mod, C, reuse_a, reuse_b)
  return __builtin_amdgcn_wmma_f32_16x16x32_bf16(false, a, false, b, (short)0, c, false, false);
}
// fast sigmoid: v_exp_f32 + v_rcp_f32 (bf16 outputs don't need IEEE division)
__device__ __forceinline__ float sigmoidf(float x) {
  return __builtin_amdgcn_rcpf(1.0f + __expf(-x));
}

#if USE_ASYNC_LDS
// one 16-byte async global->LDS move (ASYNCcnt-tracked, no VGPR data).
// Builtin signature (from toolchain diagnostic): (v4i as1*, v4i as3*, imm offset, imm cpol)
typedef __attribute__((address_space(1))) v4i* as1_v4i_p;
typedef __attribute__((address_space(3))) v4i* as3_v4i_p;
__device__ __forceinline__ void async_ld16(const __bf16* g, __bf16* l) {
  __builtin_amdgcn_global_load_async_to_lds_b128((as1_v4i_p)(g), (as3_v4i_p)(l), 0, 0);
}
#endif

// ============================================================
// Kernel 0: pack weights to bf16.
// Wbf rows: [0..127]=g_in(a) [128..255]=p_in(a) [256..383]=g_in(b)
//           [384..511]=p_in(b) [512..639]=g_out.  PoutBf: p_out rows.
// ============================================================
__global__ void k_pack(const float* __restrict__ giw, const float* __restrict__ piw,
                       const float* __restrict__ gow, const float* __restrict__ poutw,
                       __bf16* __restrict__ Wbf, __bf16* __restrict__ PoutBf) {
  int r = blockIdx.x, c = threadIdx.x;
  const float* src; __bf16* dst;
  if (r < 128)      { src = giw  + r * 128;             dst = Wbf + r * 128; }
  else if (r < 256) { src = piw  + (r - 128) * 128;     dst = Wbf + r * 128; }
  else if (r < 384) { src = giw  + (r - 256 + 128)*128; dst = Wbf + r * 128; }
  else if (r < 512) { src = piw  + (r - 384 + 128)*128; dst = Wbf + r * 128; }
  else if (r < 640) { src = gow  + (r - 512) * 128;     dst = Wbf + r * 128; }
  else              { src = poutw + (r - 640) * 128;    dst = PoutBf + (r - 640) * 128; }
  dst[c] = (__bf16)src[c];
}

// ============================================================
// Kernel 1: LayerNorm(x) + input projections via WMMA.
// Block = 256 threads (8 waves), M-tile = 64 positions.
// Writes a_t/b_t/g_t channel-major: [d][pos] bf16.
// ============================================================
#define XN_STRIDE 136   // 128 + pad: 272B rows -> 16B aligned, conflict-free b128
__global__ __launch_bounds__(256) void k1_ln_proj(
    const float* __restrict__ x, const float* __restrict__ mask,
    const float* __restrict__ nw, const float* __restrict__ nb,
    const __bf16* __restrict__ Wbf,
    __bf16* __restrict__ a_t, __bf16* __restrict__ b_t, __bf16* __restrict__ g_t) {
  __shared__ __bf16 lds_xn[64 * XN_STRIDE];
  const int t = threadIdx.x;
  const size_t posBase = (size_t)blockIdx.x * 64;

  // ---- Phase A: LayerNorm, 4 lanes per row (32 channels each) ----
  {
    int r = t >> 2, q = t & 3;
    const float4* xp = (const float4*)(x + (posBase + r) * 128 + q * 32);
    float4 xv[8];
#pragma unroll
    for (int i = 0; i < 8; ++i) xv[i] = xp[i];
    float s = 0.f, ss = 0.f;
#pragma unroll
    for (int i = 0; i < 8; ++i) {
      float4 v = xv[i];
      s  += v.x + v.y + v.z + v.w;
      ss += v.x * v.x + v.y * v.y + v.z * v.z + v.w * v.w;
    }
    s  += __shfl_xor(s, 1, 32);  s  += __shfl_xor(s, 2, 32);
    ss += __shfl_xor(ss, 1, 32); ss += __shfl_xor(ss, 2, 32);
    float m   = s * (1.0f / 128.0f);
    float inv = rsqrtf(ss * (1.0f / 128.0f) - m * m + 1e-5f);
#pragma unroll
    for (int i = 0; i < 8; ++i) {
      int k = q * 32 + i * 4;
      float4 v = xv[i];
      lds_xn[r * XN_STRIDE + k + 0] = (__bf16)((v.x - m) * inv * nw[k + 0] + nb[k + 0]);
      lds_xn[r * XN_STRIDE + k + 1] = (__bf16)((v.y - m) * inv * nw[k + 1] + nb[k + 1]);
      lds_xn[r * XN_STRIDE + k + 2] = (__bf16)((v.z - m) * inv * nw[k + 2] + nb[k + 2]);
      lds_xn[r * XN_STRIDE + k + 3] = (__bf16)((v.w - m) * inv * nw[k + 3] + nb[k + 3]);
    }
  }
  __syncthreads();

  // ---- Phase B: WMMA GEMM xn(64x128) @ W^T, fused sigmoid/gate/mask ----
  const int lane = t & 31, w = t >> 5, wm = w & 3, wn = w >> 2;
  v16bf af[4];
#pragma unroll
  for (int kk = 0; kk < 4; ++kk) af[kk] = afrag(lds_xn, 16 * wm, XN_STRIDE, 32 * kk, lane);

  const int rb = 16 * wm + ((lane & 16) ? 8 : 0);

#pragma unroll
  for (int half = 0; half < 2; ++half) {          // a-half then b-half
    __bf16* dst = half ? b_t : a_t;
    const int gbase = half * 256;
    for (int ct = wn; ct < 8; ct += 2) {          // 16-channel tiles
      const int chB = 16 * ct;
      v8f ag = {}, ap = {};
#pragma unroll
      for (int kk = 0; kk < 4; ++kk) {
        v16bf bg = bfrag(Wbf, gbase + chB,       128, 32 * kk, lane);
        v16bf bp = bfrag(Wbf, gbase + 128 + chB, 128, 32 * kk, lane);
        ag = wmma_bf16(af[kk], bg, ag);
        ap = wmma_bf16(af[kk], bp, ap);
      }
      const int ch = chB + (lane & 15);
      __bf16* cp = dst + (size_t)ch * POS + posBase;
#pragma unroll
      for (int v = 0; v < 8; ++v) {
        int row = rb + v;
        float res = sigmoidf(ag[v]) * ap[v];
        if (half == 0) res *= mask[posBase + row];
        cp[row] = (__bf16)res;
      }
    }
  }
  // g_out gate = sigmoid(xn @ g_out_w^T), stored channel-major
  for (int ct = wn; ct < 8; ct += 2) {
    const int chB = 16 * ct;
    v8f ac = {};
#pragma unroll
    for (int kk = 0; kk < 4; ++kk) {
      v16bf bg = bfrag(Wbf, 512 + chB, 128, 32 * kk, lane);
      ac = wmma_bf16(af[kk], bg, ac);
    }
    const int ch = chB + (lane & 15);
    __bf16* cp = g_t + (size_t)ch * POS + posBase;
#pragma unroll
    for (int v = 0; v < 8; ++v) cp[rb + v] = (__bf16)sigmoidf(ac[v]);
  }
}

// ============================================================
// Kernel 2: triangle einsum. Per channel d: T_d = A_d(768x768) @ B_d^T.
// Block = 256 threads (8 waves), tile 128(i) x 64(j), K panels of 32.
// Double-buffered async global->LDS when the builtins exist.
// ============================================================
#define KP 40   // LDS k-panel stride: 80B rows, 16B aligned, conflict-free
__global__ __launch_bounds__(256) void k2_triangle(
    const __bf16* __restrict__ a_t, const __bf16* __restrict__ b_t,
    float* __restrict__ tbuf) {
  __shared__ __bf16 la[2][128 * KP];
  __shared__ __bf16 lb[2][64 * KP];
  const int t = threadIdx.x, lane = t & 31, w = t >> 5;
  const int wm = w & 3, wn = w >> 2;
  const int i0 = blockIdx.x * 128, j0 = blockIdx.y * 64, d = blockIdx.z;
  const __bf16* A = a_t + (size_t)d * POS;
  const __bf16* B = b_t + (size_t)d * POS;

  v8f acc[2][2] = {};

#if USE_ASYNC_LDS
  // uniform traffic: every thread moves 3 x 16B per panel (2 for A, 1 for B)
  const int arow = t >> 1, ahf = (t & 1) * 16;
  const int brow = t >> 2, bhf = (t & 3) * 8;
  const __bf16* gA = A + (size_t)(i0 + arow) * NN + ahf;
  const __bf16* gB = B + (size_t)(j0 + brow) * NN + bhf;

  auto issue = [&](int buf, int kb) {
    const int k0 = kb * 32;
    async_ld16(gA + k0,     &la[buf][arow * KP + ahf]);
    async_ld16(gA + k0 + 8, &la[buf][arow * KP + ahf + 8]);
    async_ld16(gB + k0,     &lb[buf][brow * KP + bhf]);
  };

  issue(0, 0);
  for (int kb = 0; kb < NN / 32; ++kb) {
    const int cur = kb & 1;
    if (kb + 1 < NN / 32) {
      issue(cur ^ 1, kb + 1);
      __builtin_amdgcn_s_wait_asynccnt(3);   // in-order: cur panel resident
    } else {
      __builtin_amdgcn_s_wait_asynccnt(0);
    }
    __syncthreads();
    v16bf afr[2], bfr[2];
#pragma unroll
    for (int rt = 0; rt < 2; ++rt) afr[rt] = afrag(la[cur], 32 * wm + 16 * rt, KP, 0, lane);
#pragma unroll
    for (int ct = 0; ct < 2; ++ct) bfr[ct] = bfrag(lb[cur], 32 * wn + 16 * ct, KP, 0, lane);
#pragma unroll
    for (int rt = 0; rt < 2; ++rt)
#pragma unroll
      for (int ct = 0; ct < 2; ++ct)
        acc[rt][ct] = wmma_bf16(afr[rt], bfr[ct], acc[rt][ct]);
    __syncthreads();   // done reading cur before it is overwritten
  }
#else
  for (int kb = 0; kb < NN / 32; ++kb) {
    const int k0 = kb * 32;
    {   // cooperative panel load: each thread moves 16 bf16 (32B)
      int row = t >> 1, hf = (t & 1) * 16;
      const v8bf* sa = (const v8bf*)(A + (size_t)(i0 + row) * NN + k0 + hf);
      v8bf a0 = sa[0], a1 = sa[1];
      *(v8bf*)&la[0][row * KP + hf]     = a0;
      *(v8bf*)&la[0][row * KP + hf + 8] = a1;
      if (row < 64) {
        const v8bf* sb = (const v8bf*)(B + (size_t)(j0 + row) * NN + k0 + hf);
        v8bf b0 = sb[0], b1 = sb[1];
        *(v8bf*)&lb[0][row * KP + hf]     = b0;
        *(v8bf*)&lb[0][row * KP + hf + 8] = b1;
      }
    }
    __syncthreads();
    v16bf afr[2], bfr[2];
#pragma unroll
    for (int rt = 0; rt < 2; ++rt) afr[rt] = afrag(la[0], 32 * wm + 16 * rt, KP, 0, lane);
#pragma unroll
    for (int ct = 0; ct < 2; ++ct) bfr[ct] = bfrag(lb[0], 32 * wn + 16 * ct, KP, 0, lane);
#pragma unroll
    for (int rt = 0; rt < 2; ++rt)
#pragma unroll
      for (int ct = 0; ct < 2; ++ct)
        acc[rt][ct] = wmma_bf16(afr[rt], bfr[ct], acc[rt][ct]);
    __syncthreads();
  }
#endif

  // store T channel-major [d][i*768+j]
  float* T = tbuf + (size_t)d * POS;
  const int rbs = (lane & 16) ? 8 : 0;
#pragma unroll
  for (int rt = 0; rt < 2; ++rt)
#pragma unroll
    for (int ct = 0; ct < 2; ++ct) {
      int col = j0 + 32 * wn + 16 * ct + (lane & 15);
#pragma unroll
      for (int v = 0; v < 8; ++v) {
        int row = i0 + 32 * wm + 16 * rt + rbs + v;
        T[(size_t)row * NN + col] = acc[rt][ct][v];
      }
    }
}

// ============================================================
// Kernel 3: LayerNorm over d (gather from channel-major T) +
//           p_out projection via WMMA + sigmoid gate, fp32 out.
// ============================================================
#define LT_S 65
__global__ __launch_bounds__(256) void k3_out(
    const float* __restrict__ tbuf, const __bf16* __restrict__ g_t,
    const float* __restrict__ nw, const float* __restrict__ nb,
    const __bf16* __restrict__ PoutBf, float* __restrict__ out) {
  __shared__ float  lt[128 * LT_S];
  __shared__ float  rs[4 * 64], rq[4 * 64];
  __shared__ float  mean_s[64], inv_s[64];
  __shared__ __bf16 lds_xn[64 * XN_STRIDE];
  const int t = threadIdx.x;
  const size_t posBase = (size_t)blockIdx.x * 64;

  // ---- Phase A1: coalesced gather (lane = position), partial LN stats ----
  {
    int grp = t >> 6, pl = t & 63;
    float s = 0.f, q = 0.f;
#pragma unroll 4
    for (int cc = 0; cc < 32; ++cc) {
      int ch = grp * 32 + cc;
      float v = tbuf[(size_t)ch * POS + posBase + pl];
      lt[ch * LT_S + pl] = v;
      s += v; q += v * v;
    }
    rs[grp * 64 + pl] = s; rq[grp * 64 + pl] = q;
  }
  __syncthreads();
  if (t < 64) {
    float s = rs[t] + rs[64 + t] + rs[128 + t] + rs[192 + t];
    float q = rq[t] + rq[64 + t] + rq[128 + t] + rq[192 + t];
    float m = s * (1.0f / 128.0f);
    mean_s[t] = m;
    inv_s[t]  = rsqrtf(q * (1.0f / 128.0f) - m * m + 1e-5f);
  }
  __syncthreads();
  // ---- Phase A2: normalize into bf16 A-operand staging ----
  {
    int r = t >> 2, q3 = t & 3;
    float m = mean_s[r], inv = inv_s[r];
#pragma unroll 4
    for (int i = 0; i < 32; ++i) {
      int ch = q3 * 32 + i;
      float xn = (lt[ch * LT_S + r] - m) * inv * nw[ch] + nb[ch];
      lds_xn[r * XN_STRIDE + ch] = (__bf16)xn;
    }
  }
  __syncthreads();

  // ---- Phase B: WMMA tn(64x128) @ p_out^T, multiply by stored gate ----
  const int lane = t & 31, w = t >> 5, wm = w & 3, wn = w >> 2;
  v16bf af[4];
#pragma unroll
  for (int kk = 0; kk < 4; ++kk) af[kk] = afrag(lds_xn, 16 * wm, XN_STRIDE, 32 * kk, lane);
  const int rb = 16 * wm + ((lane & 16) ? 8 : 0);

  for (int et = wn; et < 8; et += 2) {
    v8f ac = {};
#pragma unroll
    for (int kk = 0; kk < 4; ++kk) {
      v16bf bf = bfrag(PoutBf, 16 * et, 128, 32 * kk, lane);
      ac = wmma_bf16(af[kk], bf, ac);
    }
    const int e = 16 * et + (lane & 15);
    const __bf16* gp = g_t + (size_t)e * POS + posBase;
#pragma unroll
    for (int v = 0; v < 8; ++v) {
      size_t pos = posBase + rb + v;
      float gate = (float)gp[rb + v];
      out[pos * 128 + e] = ac[v] * gate;
    }
  }
}

// ============================================================
extern "C" void kernel_launch(void* const* d_in, const int* in_sizes, int n_in,
                              void* d_out, int out_size, void* d_ws, size_t ws_size,
                              hipStream_t stream) {
  const float* x     = (const float*)d_in[0];
  const float* mask  = (const float*)d_in[1];
  const float* niw   = (const float*)d_in[2];
  const float* nib   = (const float*)d_in[3];
  const float* giw   = (const float*)d_in[4];
  const float* piw   = (const float*)d_in[5];
  const float* now_  = (const float*)d_in[6];
  const float* nob   = (const float*)d_in[7];
  const float* gow   = (const float*)d_in[8];
  const float* poutw = (const float*)d_in[9];
  float* outp = (float*)d_out;

  // workspace carve-up (~755 MB)
  __bf16* a_t    = (__bf16*)d_ws;                       // [128][POS]
  __bf16* b_t    = a_t + (size_t)DCH * POS;             // [128][POS]
  __bf16* g_t    = b_t + (size_t)DCH * POS;             // [128][POS]
  float*  tbuf   = (float*)(g_t + (size_t)DCH * POS);   // [128][POS]
  __bf16* Wbf    = (__bf16*)(tbuf + (size_t)DCH * POS); // [640][128]
  __bf16* PoutBf = Wbf + 640 * 128;                     // [128][128]

  k_pack<<<768, 128, 0, stream>>>(giw, piw, gow, poutw, Wbf, PoutBf);
  k1_ln_proj<<<(int)(POS / 64), 256, 0, stream>>>(x, mask, niw, nib, Wbf, a_t, b_t, g_t);
  k2_triangle<<<dim3(NN / 128, NN / 64, DCH), 256, 0, stream>>>(a_t, b_t, tbuf);
  k3_out<<<(int)(POS / 64), 256, 0, stream>>>(tbuf, g_t, now_, nob, PoutBf, outp);
}
